// AttentionHead_16509854286025
// MI455X (gfx1250) — compile-verified, matrix operand order checked by reading, not run on hardware
//
#include <hip/hip_runtime.h>
#include <hip/hip_bf16.h>

// Problem constants (from reference): B=4, T=4096, E=1024, H=64
#define B_  4
#define T_  4096
#define E_  1024
#define H_  64
#define BT_ (B_ * T_)
#define KS_ (E_ / 32)   // 32 K-steps of 32 over E

typedef __attribute__((ext_vector_type(16))) _Float16 v16h;
typedef __attribute__((ext_vector_type(8)))  _Float16 v8h;
typedef __attribute__((ext_vector_type(8)))  float    v8f;

union V16 { v16h v; v8h h[2]; };

// WMMA 16-bit fragment maps (ISA 7.12.2), wave32:
//  A (16xK): lanes l and l+16 both hold row M=l&15;
//    lane<16 : elems 0..7 -> K 0..7,  8..15 -> K 16..23
//    lane>=16: elems 0..7 -> K 8..15, 8..15 -> K 24..31
//  B (Kx16) mirrors A with N = lane&15.
//  C/D f32 16x16: N = lane&15, VGPR r -> M = r + 8*(lane>>4).

// -------------------------------------------------------------------------
// Prep: repack Wq|Wk|Wv (fp32 [E,H]) into f16 WMMA-B fragment order:
//   Wpk[((mat*KS_+ks)*4 + ntile)*32 + lane][0..15]  -- 32B contiguous per lane
// -------------------------------------------------------------------------
__global__ __launch_bounds__(256) void pack_w_kernel(
    const float* __restrict__ Wq, const float* __restrict__ Wk,
    const float* __restrict__ Wv, _Float16* __restrict__ Wpk)
{
  const int idx  = blockIdx.x * 256 + threadIdx.x;   // 0..12287
  const int lane = idx & 31;
  const int t    = (idx >> 5) & 3;
  const int ks   = (idx >> 7) & 31;
  const int mat  = idx >> 12;
  const int half = lane >> 4, l16 = lane & 15;
  const float* __restrict__ W = (mat == 0) ? Wq : ((mat == 1) ? Wk : Wv);
  _Float16* __restrict__ o = Wpk + (size_t)idx * 16;
  #pragma unroll
  for (int i = 0; i < 16; ++i) {
    const int k = ks * 32 + ((i >> 3) << 4) + (half << 3) + (i & 7);
    o[i] = (_Float16)W[(size_t)k * H_ + t * 16 + l16];
  }
}

// -------------------------------------------------------------------------
// QKV projection: one wave per 16-row tile computes Q, K AND V (12 f32
// accumulators).  All 12 B-fragments are preloaded per K-step (one load
// clause, independent registers) so the WMMA chain overlaps the loads.
// Q,K stored f16 [BT,H]; V stored transposed f16 [H,BT].
// -------------------------------------------------------------------------
__global__ __launch_bounds__(32) void qkv_proj_kernel(
    const float* __restrict__ x, const _Float16* __restrict__ Wpk,
    _Float16* __restrict__ qh, _Float16* __restrict__ kh,
    _Float16* __restrict__ vt)
{
  const int  tile = blockIdx.x;
  const int  lane = threadIdx.x & 31;
  const int  half = lane >> 4, l16 = lane & 15;
  const size_t row0 = (size_t)tile * 16;

  v8f acc[3][4] = {};

  #pragma unroll 1
  for (int ks = 0; ks < KS_; ++ks) {
    // A-fragment: 16x32 slice of x, fp32 -> f16 (b128 loads + pk-cvt)
    V16 a;
    const float* xp = x + (row0 + l16) * (size_t)E_ + ks * 32 + half * 8;
    #pragma unroll
    for (int g = 0; g < 2; ++g)
      #pragma unroll
      for (int i = 0; i < 8; ++i)
        a.h[g][i] = (_Float16)xp[g * 16 + i];

    // Preload all 12 B-fragments (24 x b128, independent registers)
    V16 bf[12];
    #pragma unroll
    for (int m = 0; m < 3; ++m)
      #pragma unroll
      for (int t = 0; t < 4; ++t) {
        const _Float16* wp =
            Wpk + ((((size_t)m * KS_ + ks) * 4 + t) * 32 + lane) * 16;
        bf[m * 4 + t].h[0] = *(const v8h*)(wp);
        bf[m * 4 + t].h[1] = *(const v8h*)(wp + 8);
      }

    // 12 WMMAs; loads retire progressively underneath
    #pragma unroll
    for (int m = 0; m < 3; ++m)
      #pragma unroll
      for (int t = 0; t < 4; ++t)
        acc[m][t] = __builtin_amdgcn_wmma_f32_16x16x32_f16(
            false, a.v, false, bf[m * 4 + t].v, (short)0, acc[m][t],
            false, false);
  }

  #pragma unroll
  for (int t = 0; t < 4; ++t)
    #pragma unroll
    for (int r = 0; r < 8; ++r) {
      const size_t m = row0 + r + 8 * half;
      qh[m * H_ + t * 16 + l16]            = (_Float16)acc[0][t][r];
      kh[m * H_ + t * 16 + l16]            = (_Float16)acc[1][t][r];
      vt[(size_t)(t * 16 + l16) * BT_ + m] = (_Float16)acc[2][t][r];
    }
}

// -------------------------------------------------------------------------
// Flash attention: one wave per 16 query rows.  Key blocks of 64 (4 S-tiles)
// amortize the shuffle-reduce softmax over 16 WMMAs per block.  K and V
// fragments are preloaded in bulk ahead of each WMMA chain.
// -------------------------------------------------------------------------
__global__ __launch_bounds__(32) void flash_attn_kernel(
    const _Float16* __restrict__ qh, const _Float16* __restrict__ kh,
    const _Float16* __restrict__ vt, float* __restrict__ out)
{
  __shared__ __align__(16) _Float16 Plds[16 * 64];

  const int qt   = blockIdx.x;
  const int b    = blockIdx.y;
  const int lane = threadIdx.x & 31;
  const int half = lane >> 4, l16 = lane & 15;
  const int q0   = qt * 16;
  const float scale = 0.125f;   // 1/sqrt(H=64)

  const _Float16* __restrict__ qb = qh + (size_t)b * T_ * H_;
  const _Float16* __restrict__ kb = kh + (size_t)b * T_ * H_;
  const _Float16* __restrict__ vb = vt + (size_t)b * T_;   // row h at vb[h*BT_+key]

  V16 Qa[2];
  #pragma unroll
  for (int c = 0; c < 2; ++c) {
    const _Float16* p = qb + (size_t)(q0 + l16) * H_ + c * 32 + half * 8;
    Qa[c].h[0] = *(const v8h*)(p);
    Qa[c].h[1] = *(const v8h*)(p + 16);
  }

  v8f   O[4] = {};
  float mi[8], li[8];
  #pragma unroll
  for (int r = 0; r < 8; ++r) { mi[r] = -1.0e30f; li[r] = 0.0f; }

  const int nblk = (q0 + 15) / 64 + 1;   // causal: keys 0 .. q0+15
  #pragma unroll 1
  for (int jb = 0; jb < nblk; ++jb) {
    const int k0 = jb * 64;

    if (jb + 1 < nblk) {   // prefetch next key block (global_prefetch_b8)
      __builtin_prefetch(kb + (size_t)(k0 + 64 + lane) * H_, 0, 1);
      __builtin_prefetch(vb + (size_t)(lane & (H_ - 1)) * BT_ + k0 + 64, 0, 1);
    }

    // ---- preload all 8 K-fragments (one clause), then 8 S-WMMAs ----
    V16 Kf[8];
    #pragma unroll
    for (int jj = 0; jj < 4; ++jj)
      #pragma unroll
      for (int c = 0; c < 2; ++c) {
        const _Float16* p =
            kb + (size_t)(k0 + jj * 16 + l16) * H_ + c * 32 + half * 8;
        Kf[jj * 2 + c].h[0] = *(const v8h*)(p);
        Kf[jj * 2 + c].h[1] = *(const v8h*)(p + 16);
      }

    v8f S[4];
    #pragma unroll
    for (int jj = 0; jj < 4; ++jj) {
      v8f s = {};
      #pragma unroll
      for (int c = 0; c < 2; ++c)
        s = __builtin_amdgcn_wmma_f32_16x16x32_f16(
            false, Qa[c].v, false, Kf[jj * 2 + c].v, (short)0, s,
            false, false);
      const int kcol = k0 + jj * 16 + l16;
      #pragma unroll
      for (int r = 0; r < 8; ++r) {
        const float v    = s[r] * scale;
        const int   qrow = q0 + r + 8 * half;
        S[jj][r] = (kcol > qrow) ? -1.0e30f : v;     // causal mask
      }
    }

    // ---- online softmax over 64 columns ----
    float mx[8];
    #pragma unroll
    for (int r = 0; r < 8; ++r)
      mx[r] = fmaxf(fmaxf(S[0][r], S[1][r]), fmaxf(S[2][r], S[3][r]));
    #pragma unroll
    for (int off = 1; off < 16; off <<= 1)
      #pragma unroll
      for (int r = 0; r < 8; ++r)
        mx[r] = fmaxf(mx[r], __shfl_xor(mx[r], off, 32));

    float alpha[8], rs[8];
    #pragma unroll
    for (int r = 0; r < 8; ++r) {
      const float mnew = fmaxf(mi[r], mx[r]);
      alpha[r] = __expf(mi[r] - mnew);
      mi[r]    = mnew;
      rs[r]    = 0.0f;
    }
    #pragma unroll
    for (int jj = 0; jj < 4; ++jj)
      #pragma unroll
      for (int r = 0; r < 8; ++r) {
        const float p = __expf(S[jj][r] - mi[r]);
        S[jj][r] = p;
        rs[r]   += p;
      }
    #pragma unroll
    for (int off = 1; off < 16; off <<= 1)
      #pragma unroll
      for (int r = 0; r < 8; ++r)
        rs[r] += __shfl_xor(rs[r], off, 32);
    #pragma unroll
    for (int r = 0; r < 8; ++r) li[r] = li[r] * alpha[r] + rs[r];
    #pragma unroll
    for (int t = 0; t < 4; ++t)
      #pragma unroll
      for (int r = 0; r < 8; ++r) O[t][r] *= alpha[r];

    // ---- P: C-layout -> LDS (16x64 f16) -> two 16x32 A-fragments ----
    #pragma unroll
    for (int jj = 0; jj < 4; ++jj)
      #pragma unroll
      for (int r = 0; r < 8; ++r)
        Plds[(r + 8 * half) * 64 + jj * 16 + l16] = (_Float16)S[jj][r];
    asm volatile("s_wait_dscnt 0" ::: "memory");

    V16 Pa[2];
    #pragma unroll
    for (int c = 0; c < 2; ++c) {
      Pa[c].h[0] = *(const v8h*)(Plds + l16 * 64 + c * 32 + half * 8);
      Pa[c].h[1] = *(const v8h*)(Plds + l16 * 64 + c * 32 + half * 8 + 16);
    }

    // ---- preload all 8 V-fragments, then 8 P*V WMMAs ----
    V16 Vf[8];
    #pragma unroll
    for (int c = 0; c < 2; ++c)
      #pragma unroll
      for (int t = 0; t < 4; ++t) {
        const _Float16* p =
            vb + (size_t)(t * 16 + l16) * BT_ + k0 + c * 32 + half * 8;
        Vf[c * 4 + t].h[0] = *(const v8h*)(p);
        Vf[c * 4 + t].h[1] = *(const v8h*)(p + 16);
      }
    #pragma unroll
    for (int c = 0; c < 2; ++c)
      #pragma unroll
      for (int t = 0; t < 4; ++t)
        O[t] = __builtin_amdgcn_wmma_f32_16x16x32_f16(
            false, Pa[c].v, false, Vf[c * 4 + t].v, (short)0, O[t],
            false, false);
  }

  // ---- normalize, store fp32 [B,T,H] ----
  float invl[8];
  #pragma unroll
  for (int r = 0; r < 8; ++r) invl[r] = 1.0f / li[r];
  #pragma unroll
  for (int t = 0; t < 4; ++t)
    #pragma unroll
    for (int r = 0; r < 8; ++r)
      out[((size_t)b * T_ + q0 + r + 8 * half) * H_ + t * 16 + l16] =
          O[t][r] * invl[r];
}

extern "C" void kernel_launch(void* const* d_in, const int* in_sizes, int n_in,
                              void* d_out, int out_size, void* d_ws, size_t ws_size,
                              hipStream_t stream) {
  (void)in_sizes; (void)n_in; (void)out_size; (void)ws_size;
  const float* x  = (const float*)d_in[0];
  const float* Wq = (const float*)d_in[1];
  const float* Wk = (const float*)d_in[2];
  const float* Wv = (const float*)d_in[3];
  float* out = (float*)d_out;

  // ws: q [BT,H] | k [BT,H] | v^T [H,BT] (f16, 2MB each) | packed W (384KB)
  _Float16* qh  = (_Float16*)d_ws;
  _Float16* kh  = qh + (size_t)BT_ * H_;
  _Float16* vt  = kh + (size_t)BT_ * H_;
  _Float16* Wpk = vt + (size_t)H_ * BT_;

  pack_w_kernel<<<48, 256, 0, stream>>>(Wq, Wk, Wv, Wpk);
  qkv_proj_kernel<<<BT_ / 16, 32, 0, stream>>>(x, Wpk, qh, kh, vt);
  flash_attn_kernel<<<dim3(T_ / 16, B_), 32, 0, stream>>>(qh, kh, vt, out);
}
